// HarmonicOscillator_27212912788115
// MI455X (gfx1250) — matrix operand order repaired and checked by reading, not compiled, not run on hardware
//
#include <hip/hip_runtime.h>
#include <hip/hip_bf16.h>

typedef __attribute__((ext_vector_type(16))) _Float16 v16h;
typedef __attribute__((ext_vector_type(8)))  float    v8f;

#define FRAMES 500
#define NHARM  64
#define BS     512
#define FS_D   44100.0
#define NYQ_F  22050.0f
#define TWO_PI 6.283185307179586

// ---------------------------------------------------------------------------
// Kernel 1: per-batch exclusive prefix scan (f64) of per-frame phase sums.
// Frame k sum of upsampled f0: D_k = 512*f0_k + (f0_{k+1}-f0_k)*255.5
// (closed form of sum_{i=0..511} linear interp; last frame repeats endpoint).
// ---------------------------------------------------------------------------
__global__ __launch_bounds__(512) void harm_scan_kernel(
    const float* __restrict__ f0g, double* __restrict__ P) {
  const int b = blockIdx.x;
  const int t = threadIdx.x;
  __shared__ double s[512];

  double D = 0.0;
  if (t < FRAMES) {
    double a = (double)f0g[b * FRAMES + t];
    int k1   = (t + 1 < FRAMES) ? t + 1 : FRAMES - 1;
    double c = (double)f0g[b * FRAMES + k1];
    D = 512.0 * a + (c - a) * 255.5;
  }
  s[t] = D;
  __syncthreads();
  // Hillis-Steele inclusive scan over 512 (f64)
  for (int off = 1; off < 512; off <<= 1) {
    double v = (t >= off) ? s[t - off] : 0.0;
    __syncthreads();
    s[t] += v;
    __syncthreads();
  }
  if (t < FRAMES) P[b * FRAMES + t] = (t == 0) ? 0.0 : s[t - 1];
}

// ---------------------------------------------------------------------------
// Kernel 2: one workgroup per (frame, batch). 8 waves x 4 tiles x 16 samples.
// Per tile: A(16x32 f16) = sin matrix (hw v_sin, revolutions), two chained
// v_wmma_f32_16x16x32_f16 against B holding masked amps of frame k (col 0)
// and frame k+1 (col 1); blend columns with the per-sample interp fraction.
// ---------------------------------------------------------------------------
__global__ __launch_bounds__(256) void harm_wmma_kernel(
    const float*  __restrict__ f0g,
    const float*  __restrict__ ampsg,
    const float*  __restrict__ initg,
    const double* __restrict__ P,
    float*        __restrict__ out) {
  const int frame = blockIdx.x;
  const int b     = blockIdx.y;
  const int tid   = threadIdx.x;
  const int wave  = tid >> 5;
  const int lane  = tid & 31;
  const int k1    = (frame + 1 < FRAMES) ? frame + 1 : FRAMES - 1;

  __shared__ _Float16 sA[NHARM];        // masked amps, frame k
  __shared__ _Float16 sB[NHARM];        // masked amps, frame k+1
  __shared__ float    scol[8][16][2];   // per-wave D columns 0,1

  const float f0k  = f0g[b * FRAMES + frame];
  const float f0k1 = f0g[b * FRAMES + k1];

  // Nyquist anti-alias mask applied at frame rate (matches reference: aa
  // computed per frame BEFORE upsampling), converted to f16 for WMMA.
  if (tid < NHARM) {
    float aa = ((f0k * (float)(tid + 1)) < NYQ_F ? 1.0f : 0.0f) + 1.0e-4f;
    sA[tid] = (_Float16)(ampsg[(b * FRAMES + frame) * NHARM + tid] * aa);
  } else if (tid < 2 * NHARM) {
    int h = tid - NHARM;
    float aa = ((f0k1 * (float)(h + 1)) < NYQ_F ? 1.0f : 0.0f) + 1.0e-4f;
    sB[h] = (_Float16)(ampsg[(b * FRAMES + k1) * NHARM + h] * aa);
  }
  __syncthreads();

  // --- Build B matrices (K=32 f16 B layout: lanes 0-15 -> column
  // N=lane with K in {0..7,16..23}; lanes 16-31 -> column N=lane-16 with
  // K in {8..15,24..31}). Only columns 0 (frame k) and 1 (frame k+1) are non-zero.
  const int col  = lane & 15;
  const int kofs = (lane < 16) ? 0 : 8;
  v16h B0, B1;  // harmonics 1..32 and 33..64
#pragma unroll
  for (int e = 0; e < 16; ++e) {
    int K = kofs + e + ((e >= 8) ? 8 : 0);
    _Float16 v0 = (_Float16)0.0f, v1 = (_Float16)0.0f;
    if (col == 0)      { v0 = sA[K]; v1 = sA[32 + K]; }
    else if (col == 1) { v0 = sB[K]; v1 = sB[32 + K]; }
    B0[e] = v0;
    B1[e] = v1;
  }

  const double Pk    = P[b * FRAMES + frame];
  const double initp = (double)initg[b];
  const double f0kd  = (double)f0k;
  const double slope = ((double)f0k1 - f0kd) * (1.0 / 512.0);
  const int    m     = lane & 15;   // row of this lane within the tile

#pragma unroll
  for (int p = 0; p < 4; ++p) {
    const int tile = wave * 4 + p;
    const int i    = tile * 16 + m;              // sample index in frame
    // Inclusive cumsum through sample i (closed form, f64), then mod 1.
    const double ii   = (double)i;
    const double W    = (ii + 1.0) * f0kd + slope * (ii * (ii + 1.0) * 0.5);
    const double cum  = (Pk + W) * (1.0 / FS_D) + initp;
    const double frac = cum - floor(cum);        // phase in revolutions [0,1)
    const float  fr32 = (float)frac;

    // A-matrix (16-bit 16x32 layout): lane<16 holds row M=lane, K in
    // {0..7,16..23}; lane>=16 holds row M=lane-16, K in {8..15,24..31}.
    // A[m][K] = sin(2*pi * (H+K+1) * frac) via hw v_sin (revolutions input).
    v16h A0, A1;
#pragma unroll
    for (int e = 0; e < 16; ++e) {
      int K = e + ((lane >= 16) ? 8 : 0) + ((e >= 8) ? 8 : 0);
      float h1 = (float)(K + 1);
      A0[e] = (_Float16)__builtin_amdgcn_sinf(h1 * fr32);            // h=K+1
      A1[e] = (_Float16)__builtin_amdgcn_sinf((h1 + 32.0f) * fr32);  // h=K+33
    }

    v8f c = {};
    c = __builtin_amdgcn_wmma_f32_16x16x32_f16(false, A0, false, B0,
                                               (short)0, c, false, false);
    c = __builtin_amdgcn_wmma_f32_16x16x32_f16(false, A1, false, B1,
                                               (short)0, c, false, false);

    // D layout: VGPR r, lanes 0-15 -> M=r, N=lane; lanes 16-31 -> M=8+r,
    // N=lane-16. Columns 0/1 live in lanes {0,1,16,17}; stage via LDS
    // (same-wave DS ops are in order, compiler inserts dscnt waits).
    if (lane == 0 || lane == 1 || lane == 16 || lane == 17) {
      const int rbase = (lane >= 16) ? 8 : 0;
      const int cc    = lane & 1;
#pragma unroll
      for (int r = 0; r < 8; ++r) scol[wave][rbase + r][cc] = c[r];
    }
    const float d0 = scol[wave][m][0];
    const float d1 = scol[wave][m][1];
    if (lane < 16) {
      const float fr = (float)i * (1.0f / 512.0f);   // interp fraction i/512
      out[b * (FRAMES * BS) + frame * BS + i] = d0 + fr * (d1 - d0);
      if (frame == FRAMES - 1 && i == BS - 1)        // final_phase output
        out[4 * (FRAMES * BS) + b] = (float)(TWO_PI * frac);
    }
  }
}

extern "C" void kernel_launch(void* const* d_in, const int* in_sizes, int n_in,
                              void* d_out, int out_size, void* d_ws, size_t ws_size,
                              hipStream_t stream) {
  const float* f0   = (const float*)d_in[0];   // (4, 500, 1)  f32
  const float* amps = (const float*)d_in[1];   // (4, 500, 64) f32
  const float* init = (const float*)d_in[2];   // (4, 1, 1)    f32
  float* out = (float*)d_out;                  // 4*256000 signal + 4 phase
  double* P  = (double*)d_ws;                  // 4*500 f64 frame prefixes

  harm_scan_kernel<<<dim3(4), 512, 0, stream>>>(f0, P);
  harm_wmma_kernel<<<dim3(FRAMES, 4), 256, 0, stream>>>(f0, amps, init, P, out);
}
